// GCNLayer_8057358648341
// MI455X (gfx1250) — matrix operand order repaired and checked by reading, not compiled, not run on hardware
//
#include <hip/hip_runtime.h>
#include <hip/hip_bf16.h>
#include <math.h>

typedef __attribute__((ext_vector_type(16))) __bf16        v16bf;
typedef __attribute__((ext_vector_type(8)))  float         v8f;
typedef __attribute__((ext_vector_type(4)))  float         f32x4;
typedef __attribute__((ext_vector_type(4)))  unsigned int  u32x4;

#define NND 8192
#define CIN 32
#define COUT 32

// ---------- pass 1a: zero colsum ----------
__global__ __launch_bounds__(256) void k_zero(float* __restrict__ p, int n) {
    int i = blockIdx.x * 256 + threadIdx.x;
    if (i < n) p[i] = 0.0f;
}

// ---------- pass 1b: column sums of ind = (adj != 0) ----------
// grid = (8, 64): 8 column-groups of 1024 cols (256 thr * float4), 64 row-groups of 128 rows.
// Partials are integers in f32 -> atomic order does not affect the result (deterministic).
__global__ __launch_bounds__(256) void k_colsum(const float* __restrict__ adj,
                                                float* __restrict__ colsum) {
    const int j4 = (blockIdx.x * 256 + threadIdx.x) * 4;
    const int r0 = blockIdx.y * 128;
    f32x4 s = {0.f, 0.f, 0.f, 0.f};
    for (int r = 0; r < 128; ++r) {
        f32x4 v = __builtin_nontemporal_load(
            (const f32x4*)(adj + (size_t)(r0 + r) * NND + j4));
        s.x += (v.x != 0.f) ? 1.f : 0.f;
        s.y += (v.y != 0.f) ? 1.f : 0.f;
        s.z += (v.z != 0.f) ? 1.f : 0.f;
        s.w += (v.w != 0.f) ? 1.f : 0.f;
    }
    atomicAdd(&colsum[j4 + 0], s.x);
    atomicAdd(&colsum[j4 + 1], s.y);
    atomicAdd(&colsum[j4 + 2], s.z);
    atomicAdd(&colsum[j4 + 3], s.w);
}

// ---------- pass 2: dis[j] = deg^-0.5, c1[j] = 1 - 2*ind[j][j] ----------
__global__ __launch_bounds__(256) void k_norm(const float* __restrict__ adj,
                                              const float* __restrict__ colsum,
                                              float* __restrict__ dis,
                                              float* __restrict__ c1) {
    int j = blockIdx.x * 256 + threadIdx.x;
    if (j >= NND) return;
    float indjj = (adj[(size_t)j * NND + j] != 0.f) ? 1.f : 0.f;
    float deg = 2.f * colsum[j] + 1.f - 2.f * indjj;  // == colsum of A
    dis[j] = rsqrtf(deg);
    c1[j]  = 1.f - 2.f * indjj;
}

// ---------- pass 3: zT[d][i] = bf16( dis[i] * (x@W)[i][d] ), row-major 32 x 8192 ----------
__global__ __launch_bounds__(256) void k_xw(const float* __restrict__ x,
                                            const float* __restrict__ W,
                                            const float* __restrict__ dis,
                                            __bf16* __restrict__ zT) {
    const int lane = threadIdx.x & 31;                // lane = output channel d
    const int i = blockIdx.x * 8 + (threadIdx.x >> 5); // one wave per row i
    const float* xr = x + (size_t)i * CIN;
    float acc = 0.f;
#pragma unroll
    for (int c = 0; c < CIN; ++c) acc += xr[c] * W[c * COUT + lane];
    zT[(size_t)lane * NND + i] = (__bf16)(dis[i] * acc);
}

// ---------- pass 4: S^T = zT x ind via v_wmma_f32_16x16x32_bf16, fused epilogue ----------
// Wave w of block b owns j0 = (b*8+w)*16; accumulates D tiles for d in [0,16) and [16,32).
// A (16x32 bf16): lane m=lane&15, K = {kb..kb+7, kb+16..kb+23}, kb = lane<16 ? 0 : 8.
// B (32x16 bf16): lane N=lane&15, K = {kb..kb+7, kb+16..kb+23}: adj f32 at col j0+N, binarized.
// D (16x16 f32):  VGPR v, lane: M = v + 8*(lane>=16), N = lane&15.
__global__ __launch_bounds__(256) void k_gcn_wmma(const float* __restrict__ adj,
                                                  const __bf16* __restrict__ zT,
                                                  const float* __restrict__ dis,
                                                  const float* __restrict__ c1,
                                                  const float* __restrict__ bias,
                                                  float* __restrict__ out) {
    const int lane = threadIdx.x & 31;
    const int wave = threadIdx.x >> 5;
    const int j0 = (blockIdx.x * 8 + wave) * 16;
    const int ml = lane & 15;
    const int kb = (lane < 16) ? 0 : 8;
    const int hi = (lane >= 16) ? 8 : 0;

    const __bf16* zr0 = zT + (size_t)ml * NND;         // rows d = 0..15
    const __bf16* zr1 = zT + (size_t)(16 + ml) * NND;  // rows d = 16..31
    const float*  brow = adj + j0 + ml; // col j0+ml

    v8f acc0 = {};
    v8f acc1 = {};

#pragma unroll 2
    for (int i0 = 0; i0 < NND; i0 += 32) {
        // A tiles from zT (bf16, L2-resident): two 16B contiguous chunks per lane
        union { v16bf v; u32x4 u[2]; } a0, a1;
        a0.u[0] = *(const u32x4*)(zr0 + i0 + kb);
        a0.u[1] = *(const u32x4*)(zr0 + i0 + kb + 16);
        a1.u[0] = *(const u32x4*)(zr1 + i0 + kb);
        a1.u[1] = *(const u32x4*)(zr1 + i0 + kb + 16);

        // B tile: stream 16 f32 of adj (nontemporal), binarize -> bf16
        const float* bp = brow + (size_t)i0 * NND;
        v16bf bt;
#pragma unroll
        for (int s = 0; s < 16; ++s) {
            const int k = (s < 8) ? (kb + s) : (kb + 8 + s);
            float q = __builtin_nontemporal_load(bp + (size_t)k * NND);
            bt[s] = (__bf16)((q != 0.f) ? 1.f : 0.f);
        }

        acc0 = __builtin_amdgcn_wmma_f32_16x16x32_bf16(
            false, a0.v, false, bt, (short)0, acc0, false, false);
        acc1 = __builtin_amdgcn_wmma_f32_16x16x32_bf16(
            false, a1.v, false, bt, (short)0, acc1, false, false);
    }

    // Epilogue: h = dis[j]*(2*S + c1[j]*z[j][d]) + b[d]; out[d][j] = tanh(h)
    const int   n    = j0 + ml;
    const float disn = dis[n];
    const float c1n  = c1[n];
#pragma unroll
    for (int v = 0; v < 8; ++v) {
        const int m0 = v + hi;       // d for acc0 tile
        const int m1 = 16 + v + hi;  // d for acc1 tile
        float z0 = (float)zT[(size_t)m0 * NND + n];
        float z1 = (float)zT[(size_t)m1 * NND + n];
        float h0 = disn * (2.f * acc0[v] + c1n * z0) + bias[m0];
        float h1 = disn * (2.f * acc1[v] + c1n * z1) + bias[m1];
        out[(size_t)m0 * NND + n] = tanhf(h0);
        out[(size_t)m1 * NND + n] = tanhf(h1);
    }
}

extern "C" void kernel_launch(void* const* d_in, const int* in_sizes, int n_in,
                              void* d_out, int out_size, void* d_ws, size_t ws_size,
                              hipStream_t stream) {
    const float* x   = (const float*)d_in[0];   // [8192, 32]
    const float* adj = (const float*)d_in[1];   // [8192, 8192]
    const float* W   = (const float*)d_in[2];   // [32, 32]
    const float* b   = (const float*)d_in[3];   // [32]
    float* out = (float*)d_out;                 // [32, 8192]

    // workspace layout: colsum | dis | c1 | zT(bf16 32x8192)  ~= 608 KB
    float*  colsum = (float*)d_ws;
    float*  dis    = colsum + NND;
    float*  c1     = dis + NND;
    __bf16* zT     = (__bf16*)(c1 + NND);

    k_zero<<<NND / 256, 256, 0, stream>>>(colsum, NND);
    k_colsum<<<dim3(8, 64), 256, 0, stream>>>(adj, colsum);
    k_norm<<<NND / 256, 256, 0, stream>>>(adj, colsum, dis, c1);
    k_xw<<<NND / 8, 256, 0, stream>>>(x, W, dis, zT);
    k_gcn_wmma<<<NND / (8 * 16), 256, 0, stream>>>(adj, zT, dis, c1, b, out);
}
